// DynamicGraphNN_50130858279696
// MI455X (gfx1250) — matrix operand order repaired and checked by reading, not compiled, not run on hardware
//
#include <hip/hip_runtime.h>

// Problem constants (match reference)
#define T_STEPS 20
#define BATCH   32
#define NNODES  100
#define DINF    2
#define HID     128
#define DOUTF   2
#define BNN     (BATCH * NNODES)   // 3200
#define H3      (3 * HID)          // 384

typedef __attribute__((ext_vector_type(16))) _Float16 v16h;
typedef __attribute__((ext_vector_type(2)))  _Float16 v2h;
typedef __attribute__((ext_vector_type(8)))  float    v8f;

#define AT_LD 40    // 32 + 8 pad (halves) for row-major A tiles

// ---------------- gfx1250 async global->LDS copy (ASYNCcnt path) ----------------
__device__ __forceinline__ uint32_t lds_off(const void* p) {
    // generic pointer to LDS: low 32 bits are the LDS byte offset
    return (uint32_t)(uintptr_t)p;
}
__device__ __forceinline__ void async_b128(uint32_t lds_byte, const void* gaddr) {
    asm volatile("global_load_async_to_lds_b128 %0, %1, off"
                 :: "v"(lds_byte), "v"(gaddr) : "memory");
}
__device__ __forceinline__ void async_wait0() {
    asm volatile("s_wait_asynccnt 0x0" ::: "memory");
}
__device__ __forceinline__ void async_wait2() {   // in-order: <=2 pending => older done
    asm volatile("s_wait_asynccnt 0x2" ::: "memory");
}

// ---------------- WMMA fragment loaders (contiguous k = 16*h + e, same order for A and B) ----------------
// A fragment: 16(M) x 32(K) f16 tile, row-major (stride ld halves).
// Lane L: M=L%16, h=L/16: halves {h*16..h*16+15} -> 2x b128.
__device__ __forceinline__ v16h frag_a_16x32(const _Float16* base, int ld) {
    const int lane = threadIdx.x & 31;
    const int m  = lane & 15;
    const int hh = lane >> 4;
    v16h a;
#pragma unroll
    for (int e = 0; e < 16; ++e) {
        const int v  = e >> 1;
        const int kp = hh * 16 + 2 * v;
        a[e] = base[m * ld + kp + (e & 1)];
    }
    return a;
}

// B fragment from fragment-major storage: subtile = 512 halves = 32 lanes x 16
// contiguous halves per lane (lanes 0-15: K0-15, N=lane; lanes 16-31: K16-31).
__device__ __forceinline__ v16h frag_b_fragmajor(const _Float16* subtile_base) {
    const int lane = threadIdx.x & 31;
    return *(const v16h*)(subtile_base + lane * 16);   // 32B -> 2x ds_load_b128
}

// Fragment-major global index for B element (k within 32-slab, n within 16-subtile):
//   halfindex = (subtile)*512 + ((k>>4)*16 + (n&15))*16 + (k&15)

// ---------------- small kernels ----------------
// Weights -> f16, pre-swizzled fragment-major: [kslab 0..3][subtile 0..23][512]
__global__ __launch_bounds__(256) void convert_weights_kernel(
    const float* __restrict__ Wih, const float* __restrict__ Whh,
    _Float16* __restrict__ WihSw, _Float16* __restrict__ WhhSw) {
    int idx = blockIdx.x * 256 + threadIdx.x;          // 3072 threads
    if (idx >= (HID / 32) * (H3 / 16) * 32) return;
    const int kslab = idx / 768;                       // 768 = 24 subtiles * 32 lanes
    const int rem   = idx - kslab * 768;
    const int s     = rem >> 5;
    const int L     = rem & 31;
    const int n     = s * 16 + (L & 15);
    const int kbase = kslab * 32 + (L >> 4) * 16;
    v16h bi, bh;
#pragma unroll
    for (int e = 0; e < 16; ++e) {
        bi[e] = (_Float16)Wih[(size_t)(kbase + e) * H3 + n];
        bh[e] = (_Float16)Whh[(size_t)(kbase + e) * H3 + n];
    }
    *(v16h*)(WihSw + (size_t)idx * 16) = bi;   // idx*16 == ((kslab*24+s)*32+L)*16
    *(v16h*)(WhhSw + (size_t)idx * 16) = bh;
}

__global__ __launch_bounds__(256) void zero_h_kernel(float* __restrict__ Hf,
                                                     _Float16* __restrict__ Hh) {
    int idx = blockIdx.x * 256 + threadIdx.x;
    if (idx < BNN * HID) { Hf[idx] = 0.0f; Hh[idx] = (_Float16)0.0f; }
}

__global__ __launch_bounds__(256) void zero_colsum_kernel(int* __restrict__ colsum) {
    int i = blockIdx.x * 256 + threadIdx.x;
    if (i < BNN) colsum[i] = 0;
}

// deg[i] (pre-self-loop) = sum_j (A[j,i]!=0)*mask[j]; integer atomics -> deterministic
__global__ __launch_bounds__(256) void deg_partial_kernel(
    const int* __restrict__ A_t, const unsigned char* __restrict__ mask_t,
    int* __restrict__ colsum) {
    const int col = blockIdx.x * 256 + threadIdx.x;
    if (col >= BNN) return;
    const int j0 = blockIdx.y * (BNN / 8);
    int s = 0;
    for (int j = j0; j < j0 + (BNN / 8); ++j) {
        s += (A_t[(size_t)j * BNN + col] != 0) & (mask_t[j] != 0);
    }
    atomicAdd(&colsum[col], s);
}

__global__ __launch_bounds__(256) void dinv_kernel(
    const int* __restrict__ colsum, const unsigned char* __restrict__ mask_t,
    float* __restrict__ dinv) {
    int i = blockIdx.x * 256 + threadIdx.x;
    if (i < BNN) {
        dinv[i] = (mask_t[i] != 0) ? rsqrtf((float)(colsum[i] + 1)) : 0.0f;
    }
}

// Usw = dinv * (x_t @ W_gcn) in fragment-major f16: [kslab 0..99][subtile 0..7][512].
// Thread handles 16 consecutive k (nodes), fixed n (feature).
__global__ __launch_bounds__(256) void xw_kernel(
    const float* __restrict__ x_t, const float* __restrict__ Wg,
    const float* __restrict__ dinv, _Float16* __restrict__ Usw) {
    int idx = blockIdx.x * 256 + threadIdx.x;          // 25600 threads
    if (idx >= (BNN / 16) * HID) return;
    const int kb = idx >> 7;                           // 0..199 (16-row group)
    const int n  = idx & 127;
    const float w0 = Wg[n], w1 = Wg[HID + n];
    const float* xp = x_t + (size_t)kb * 32;
    const float* dp = dinv + (size_t)kb * 16;
    v16h buf;
#pragma unroll
    for (int e = 0; e < 16; ++e) {
        float v = xp[2 * e] * w0 + xp[2 * e + 1] * w1;
        buf[e] = (_Float16)(dp[e] * v);
    }
    _Float16* dst = Usw + (size_t)(kb >> 1) * 4096 + (n >> 4) * 512
                    + ((kb & 1) * 16 + (n & 15)) * 16;
    *(v16h*)dst = buf;                                  // 32B contiguous
}

// ---------------- big WMMA kernel: AGG = (A_t + I)^T @ U, fused GCN epilogue ----------------
// 256 threads (8 waves, 4Mx2N), output tile 64(M) x 128(N), K loop over 3200.
// Double-buffered: slab k+1 async B copy + A int loads issued before computing slab k.
// Self-loop (diag) folded into the A-tile conversion: values 0/1/2 (exact in f16).
__global__ __launch_bounds__(256) void agg_wmma_kernel(
    const int* __restrict__ A_t,            // [BN][BN] (0/1)
    const _Float16* __restrict__ Usw,       // fragment-major [100][8][512]
    const float* __restrict__ dinv,         // [BN]
    const float* __restrict__ b_gcn,        // [H]
    _Float16* __restrict__ Gh) {            // [BN][H]
    __shared__ _Float16 At[2][64 * AT_LD];  // row-major A^T tiles (double buffer)
    __shared__ _Float16 Bs[2][8 * 512];     // fragment-major B slabs (double buffer)
    const int tid  = threadIdx.x;
    const int wave = tid >> 5, wm = wave & 3, wn = wave >> 2;
    const int i0   = blockIdx.x * 64;
    // A^T staging: each thread owns rows k-pair (2*kk2, 2*kk2+1) x 4 columns
    const int kk2 = tid >> 4;            // 0..15
    const int ib  = (tid & 15) * 4;      // 0..60
    v8f acc[4] = {};
    uint4 ra0, ra1;                      // prefetched A ints (2 rows x 4)

    auto issueB = [&](int k0, int p) {
        const _Float16* src = Usw + (size_t)(k0 >> 5) * 4096 + tid * 16;
        const uint32_t l = lds_off(&Bs[p][0]) + tid * 32;
        async_b128(l, src);
        async_b128(l + 16, src + 8);
    };
    auto loadA = [&](int k0) {
        const int* r0 = A_t + (size_t)(k0 + 2 * kk2) * BNN + i0 + ib;
        ra0 = *(const uint4*)r0;
        ra1 = *(const uint4*)(r0 + BNN);
    };
    auto storeA = [&](int p, int k0) {   // pack (k,k+1) pairs -> ds_store_b32
        const uint32_t a0[4] = {ra0.x, ra0.y, ra0.z, ra0.w};
        const uint32_t a1[4] = {ra1.x, ra1.y, ra1.z, ra1.w};
        const int kg0 = k0 + 2 * kk2;    // global k of row 0 (row 1 = kg0+1)
#pragma unroll
        for (int e = 0; e < 4; ++e) {
            const int cg = i0 + ib + e;  // global column (output row i)
            float f0 = (a0[e] ? 1.0f : 0.0f) + ((kg0     == cg) ? 1.0f : 0.0f);
            float f1 = (a1[e] ? 1.0f : 0.0f) + ((kg0 + 1 == cg) ? 1.0f : 0.0f);
            v2h pv;
            pv[0] = (_Float16)f0;
            pv[1] = (_Float16)f1;
            *(v2h*)(&At[p][(ib + e) * AT_LD + 2 * kk2]) = pv;
        }
    };

    // prologue: stage slab 0 into buffer 0
    issueB(0, 0);
    loadA(0);
    storeA(0, 0);
    int p = 0;
    for (int k0 = 0; k0 < BNN; k0 += 32) {
        const bool hasNext = (k0 + 32) < BNN;
        if (hasNext) {
            issueB(k0 + 32, p ^ 1);      // overlap copies with this slab's WMMAs
            loadA(k0 + 32);              // sync loads, consumed after compute
            if (k0 + 64 < BNN)           // warm L2 one more slab ahead
                __builtin_prefetch(A_t + (size_t)(k0 + 64 + 2 * kk2) * BNN + i0 + ib, 0, 1);
            async_wait2();               // in-order: buffer p's 2 copies complete
        } else {
            async_wait0();
        }
        __syncthreads();                 // single barrier per K-iteration
        v16h a = frag_a_16x32(&At[p][(wm * 16) * AT_LD], AT_LD);
        v16h bfr[4];
#pragma unroll
        for (int ns = 0; ns < 4; ++ns)   // batch the ds loads, then the WMMA chain
            bfr[ns] = frag_b_fragmajor(&Bs[p][(wn * 4 + ns) * 512]);
#pragma unroll
        for (int ns = 0; ns < 4; ++ns)
            acc[ns] = __builtin_amdgcn_wmma_f32_16x16x32_f16(
                false, a, false, bfr[ns], (short)0, acc[ns], false, false);
        if (hasNext) storeA(p ^ 1, k0 + 32);  // fill other buffer; next barrier publishes
        p ^= 1;
    }
    // epilogue: g = relu( dinv[i]*acc + b_gcn )   (self-loop already in the GEMM)
    const int lane = tid & 31, nl = lane & 15, hh = lane >> 4;
#pragma unroll
    for (int ns = 0; ns < 4; ++ns) {
        const int col = wn * 64 + ns * 16 + nl;
#pragma unroll
        for (int r = 0; r < 8; ++r) {
            const int row = i0 + wm * 16 + hh * 8 + r;
            float v = dinv[row] * acc[ns][r] + b_gcn[col];
            Gh[(size_t)row * HID + col] = (_Float16)fmaxf(v, 0.0f);
        }
    }
}

// ---------------- fused dual GRU GEMM: GI = g@W_ih + b_ih, GH = h@W_hh + b_hh ----------------
// Block tile 64(M) x 128(N); grid (50, 3); K = 128 (4 iters). All staging async.
__global__ __launch_bounds__(256) void gru_gemm_kernel(
    const _Float16* __restrict__ Gh, const _Float16* __restrict__ Hh,
    const _Float16* __restrict__ WihSw, const _Float16* __restrict__ WhhSw,
    const float* __restrict__ b_ih, const float* __restrict__ b_hh,
    float* __restrict__ GI, float* __restrict__ GH) {
    __shared__ _Float16 Gt[64 * AT_LD];
    __shared__ _Float16 Ht[64 * AT_LD];
    __shared__ _Float16 BiS[8 * 512];
    __shared__ _Float16 BhS[8 * 512];
    const int tid  = threadIdx.x;
    const int wave = tid >> 5, wm = wave & 3, wn = wave >> 2;
    const int i0 = blockIdx.x * 64;
    const int n0 = blockIdx.y * 128;
    const uint32_t gtBase = lds_off(&Gt[0]);
    const uint32_t htBase = lds_off(&Ht[0]);
    const uint32_t biBase = lds_off(&BiS[0]);
    const uint32_t bhBase = lds_off(&BhS[0]);
    v8f ai[4] = {}, ah[4] = {};

    for (int k0 = 0; k0 < HID; k0 += 32) {
        {   // activation tiles [64][32] f16 row-major: contiguous 16B chunks
            const int ii = tid >> 2, part = tid & 3;
            const size_t so = (size_t)(i0 + ii) * HID + k0 + part * 8;
            const uint32_t lo = ii * (AT_LD * 2) + part * 16;
            async_b128(gtBase + lo, Gh + so);
            async_b128(htBase + lo, Hh + so);
        }
        {   // weight slabs 32x128, fragment-major contiguous 8KB each
            const size_t so = ((size_t)(k0 >> 5) * 24 + (n0 >> 4)) * 512 + tid * 16;
            const uint32_t lo = tid * 32;
            async_b128(biBase + lo, WihSw + so);
            async_b128(biBase + lo + 16, WihSw + so + 8);
            async_b128(bhBase + lo, WhhSw + so);
            async_b128(bhBase + lo + 16, WhhSw + so + 8);
        }
        async_wait0();
        __syncthreads();
        v16h ag  = frag_a_16x32(&Gt[(wm * 16) * AT_LD], AT_LD);
        v16h ahf = frag_a_16x32(&Ht[(wm * 16) * AT_LD], AT_LD);
#pragma unroll
        for (int ns = 0; ns < 4; ++ns) {
            v16h bi = frag_b_fragmajor(&BiS[(wn * 4 + ns) * 512]);
            ai[ns] = __builtin_amdgcn_wmma_f32_16x16x32_f16(
                false, ag, false, bi, (short)0, ai[ns], false, false);
            v16h bh = frag_b_fragmajor(&BhS[(wn * 4 + ns) * 512]);
            ah[ns] = __builtin_amdgcn_wmma_f32_16x16x32_f16(
                false, ahf, false, bh, (short)0, ah[ns], false, false);
        }
        __syncthreads();
    }
    const int lane = tid & 31, nl = lane & 15, hh = lane >> 4;
#pragma unroll
    for (int ns = 0; ns < 4; ++ns) {
        const int col = n0 + wn * 64 + ns * 16 + nl;
#pragma unroll
        for (int r = 0; r < 8; ++r) {
            const int row = i0 + wm * 16 + hh * 8 + r;
            GI[(size_t)row * H3 + col] = ai[ns][r] + b_ih[col];
            GH[(size_t)row * H3 + col] = ah[ns][r] + b_hh[col];
        }
    }
}

// ---------------- GRU gates + masked scatter-back ----------------
__global__ __launch_bounds__(256) void gates_kernel(
    const float* __restrict__ GI, const float* __restrict__ GH,
    const unsigned char* __restrict__ mask_t,
    float* __restrict__ Hf, _Float16* __restrict__ Hh) {
    int idx = blockIdx.x * 256 + threadIdx.x;
    if (idx >= BNN * HID) return;
    const int i = idx >> 7, c = idx & 127;
    const size_t b = (size_t)i * H3 + c;
    float ir = GI[b], iz = GI[b + HID], ig = GI[b + 2 * HID];
    float hr = GH[b], hz = GH[b + HID], hg = GH[b + 2 * HID];
    float r = 1.0f / (1.0f + __expf(-(ir + hr)));
    float z = 1.0f / (1.0f + __expf(-(iz + hz)));
    float n = tanhf(ig + r * hg);
    float hold = Hf[idx];
    float hnew = (1.0f - z) * n + z * hold;
    float hv   = (mask_t[i] != 0) ? hnew : hold;
    Hf[idx] = hv;
    Hh[idx] = (_Float16)hv;
}

// ---------------- output projection, written straight into [B,N,T,DOUT] ----------------
__global__ __launch_bounds__(256) void out_kernel(
    const float* __restrict__ Hf, const float* __restrict__ Wfc,
    const float* __restrict__ bfc, float* __restrict__ out, int t) {
    int idx = blockIdx.x * 256 + threadIdx.x;
    if (idx >= BNN * DOUTF) return;
    const int i = idx >> 1, d = idx & 1;
    float acc = bfc[d];
    const float* h = Hf + (size_t)i * HID;
#pragma unroll 8
    for (int k = 0; k < HID; ++k) acc += h[k] * Wfc[k * DOUTF + d];
    out[(size_t)i * T_STEPS * DOUTF + t * DOUTF + d] = acc;
}

// ---------------- host ----------------
extern "C" void kernel_launch(void* const* d_in, const int* in_sizes, int n_in,
                              void* d_out, int out_size, void* d_ws, size_t ws_size,
                              hipStream_t stream) {
    const float* x      = (const float*)d_in[0];
    const int*   A      = (const int*)d_in[1];
    const unsigned char* mask = (const unsigned char*)d_in[2]; // jax bool -> 1 byte
    const float* W_gcn  = (const float*)d_in[4];
    const float* b_gcn  = (const float*)d_in[5];
    const float* W_ih   = (const float*)d_in[6];
    const float* b_ih   = (const float*)d_in[7];
    const float* W_hh   = (const float*)d_in[8];
    const float* b_hh   = (const float*)d_in[9];
    const float* W_fc   = (const float*)d_in[10];
    const float* b_fc   = (const float*)d_in[11];
    float* out = (float*)d_out;

    // workspace carve-up (256B aligned)
    char* ws = (char*)d_ws;
    size_t off = 0;
    auto alloc = [&](size_t bytes) {
        char* p = ws + off;
        off += (bytes + 255) & ~(size_t)255;
        return p;
    };
    int*       colsum = (int*)      alloc(BNN * sizeof(int));
    float*     dinv   = (float*)    alloc(BNN * sizeof(float));
    _Float16*  Usw    = (_Float16*) alloc((size_t)BNN * HID * sizeof(_Float16));
    _Float16*  Gh     = (_Float16*) alloc((size_t)BNN * HID * sizeof(_Float16));
    _Float16*  Hh     = (_Float16*) alloc((size_t)BNN * HID * sizeof(_Float16));
    float*     Hf     = (float*)    alloc((size_t)BNN * HID * sizeof(float));
    float*     GI     = (float*)    alloc((size_t)BNN * H3 * sizeof(float));
    float*     GH     = (float*)    alloc((size_t)BNN * H3 * sizeof(float));
    _Float16*  WihSw  = (_Float16*) alloc((size_t)HID * H3 * sizeof(_Float16));
    _Float16*  WhhSw  = (_Float16*) alloc((size_t)HID * H3 * sizeof(_Float16));

    convert_weights_kernel<<<(3072 + 255) / 256, 256, 0, stream>>>(W_ih, W_hh, WihSw, WhhSw);
    zero_h_kernel<<<(BNN * HID + 255) / 256, 256, 0, stream>>>(Hf, Hh);

    for (int t = 0; t < T_STEPS; ++t) {
        const int*   A_t    = A + (size_t)t * BNN * BNN;
        const unsigned char* mask_t = mask + (size_t)t * BNN;
        const float* x_t    = x + (size_t)t * BNN * DINF;

        zero_colsum_kernel<<<(BNN + 255) / 256, 256, 0, stream>>>(colsum);
        deg_partial_kernel<<<dim3((BNN + 255) / 256, 8), 256, 0, stream>>>(A_t, mask_t, colsum);
        dinv_kernel<<<(BNN + 255) / 256, 256, 0, stream>>>(colsum, mask_t, dinv);
        xw_kernel<<<((BNN / 16) * HID + 255) / 256, 256, 0, stream>>>(x_t, W_gcn, dinv, Usw);
        agg_wmma_kernel<<<BNN / 64, 256, 0, stream>>>(A_t, Usw, dinv, b_gcn, Gh);
        gru_gemm_kernel<<<dim3(BNN / 64, H3 / 128), 256, 0, stream>>>(
            Gh, Hh, WihSw, WhhSw, b_ih, b_hh, GI, GH);
        gates_kernel<<<(BNN * HID + 255) / 256, 256, 0, stream>>>(GI, GH, mask_t, Hf, Hh);
        out_kernel<<<(BNN * DOUTF + 255) / 256, 256, 0, stream>>>(Hf, W_fc, b_fc, out, t);
    }
    (void)in_sizes; (void)n_in; (void)out_size; (void)ws_size;
}
